// MLP_VSA_Layer_47622597378277
// MI455X (gfx1250) — matrix operand order repaired and checked by reading, not compiled, not run on hardware
//
#include <hip/hip_runtime.h>
#include <math.h>

// ---------------------------------------------------------------------------
// Types for CDNA5 WMMA (wave32, v_wmma_f32_16x16x32_bf16) + TDM descriptors
// ---------------------------------------------------------------------------
typedef __bf16 bf16;
typedef __attribute__((ext_vector_type(16))) __bf16 v16bf;
typedef __attribute__((ext_vector_type(8)))  __bf16 v8bf;
typedef __attribute__((ext_vector_type(8)))  float  v8f;
typedef __attribute__((ext_vector_type(4)))  unsigned uint32x4;
typedef __attribute__((ext_vector_type(8)))  int      int32x8;
typedef __attribute__((ext_vector_type(4)))  int      int32x4;

#define DEV __device__ __forceinline__

DEV bf16 f2bf(float f) {
  unsigned u = __builtin_bit_cast(unsigned, f);
  unsigned r = u + 0x7FFFu + ((u >> 16) & 1u);   // round-to-nearest-even
  unsigned short h = (unsigned short)(r >> 16);
  return __builtin_bit_cast(bf16, h);
}

DEV unsigned pack2bf(float lo, float hi) {
  unsigned a = (unsigned)__builtin_bit_cast(unsigned short, f2bf(lo));
  unsigned b = (unsigned)__builtin_bit_cast(unsigned short, f2bf(hi));
  return a | (b << 16);
}

// ordered-uint mapping so unsigned atomicMax == float max
DEV unsigned map_ord(float f) {
  unsigned u = __builtin_bit_cast(unsigned, f);
  return (u & 0x80000000u) ? ~u : (u | 0x80000000u);
}
DEV float unmap_ord(unsigned u) {
  unsigned v = (u & 0x80000000u) ? (u & 0x7FFFFFFFu) : ~u;
  return __builtin_bit_cast(float, v);
}

// ---------------------------------------------------------------------------
// WMMA GEMM:  C[Mr,Nout] = act( A[Mr,Kd] @ W[Nout,Kd]^T + bias )  (+= if ACCUM)
// Block tile 64x64, 8 waves (wave32); each wave computes a 16x32 C tile with
// two v_wmma_f32_16x16x32_bf16 per 32-wide K step.
//   - W tile (64 rows x 32 k) DMA'd by the Tensor Data Mover into LDS (f32),
//     with descriptor-based OOB zeroing (no W-side guards at all), then
//     cooperatively packed to bf16.
//   - A tile staged with vectorized float4 loads on the interior fast path.
// ---------------------------------------------------------------------------
template <int RELU, int ACCUM>
__global__ __launch_bounds__(256) void k_gemm(
    const float* __restrict__ A, const float* __restrict__ W,
    const float* __restrict__ bias, float* __restrict__ C,
    int Mr, int Nout, int Kd) {
  __shared__ bf16  As[64][40];   // row stride 80B -> 16B-aligned fragments
  __shared__ bf16  Bs[64][40];
  __shared__ float Ws[64][32];   // raw f32 W tile written by TDM

  const int tid  = threadIdx.x;
  const int lane = tid & 31;
  const int wave = tid >> 5;     // 0..7
  const int wm   = wave & 3;     // 4 M sub-tiles
  const int wn   = wave >> 2;    // 2 N sub-tiles (32 cols each)
  const int m0   = blockIdx.y * 64;
  const int n0   = blockIdx.x * 64;

  const int lrow = tid >> 2;        // 0..63 (tile row for cooperative fill)
  const int lcol = (tid & 3) * 8;   // 0,8,16,24

  const bool wave0 = (wave == 0);
  const unsigned ldsWs = (unsigned)(unsigned long long)(&Ws[0][0]);

  v8f acc0 = {};
  v8f acc1 = {};

  for (int k0 = 0; k0 < Kd; k0 += 32) {
    // ---- (1) wave0 issues TDM load of W tile [n0,n0+64) x [k0,k0+32) -> Ws
    if (wave0) {
      unsigned long long ga =
          (unsigned long long)(const void*)(W + (size_t)n0 * Kd + k0);
      int td0 = Kd - k0;   if (td0 < 0) td0 = 0;   // remaining elems dim0 (OOB->0)
      int td1 = Nout - n0; if (td1 < 0) td1 = 0;   // remaining rows dim1 (OOB->0)
      uint32x4 g0;
      g0[0] = 1u;                                        // count=1, is_restore=0
      g0[1] = ldsWs;                                     // lds_addr (bytes)
      g0[2] = (unsigned)ga;                              // global_addr[31:0]
      g0[3] = ((unsigned)(ga >> 32) & 0x01FFFFFFu)       // global_addr[56:32]
              | 0x80000000u;                             // type=2 (bits 127:126)
      int32x8 gd;
      gd[0] = 0x00020000;                                // data_size=2 (4B), no mask
      gd[1] = (int)(((unsigned)td0 & 0xFFFFu) << 16);    // tensor_dim0[15:0]
      gd[2] = (int)(((unsigned)td0 >> 16) |
                    (((unsigned)td1 & 0xFFFFu) << 16));  // td0[31:16] | td1[15:0]
      gd[3] = (int)(((unsigned)td1 >> 16) | (32u << 16));// td1[31:16] | tile_dim0=32
      gd[4] = 64;                                        // tile_dim1=64, tile_dim2=0
      gd[5] = Kd;                                        // tensor_dim0_stride[31:0]
      gd[6] = 0;
      gd[7] = 0;
      int32x4 z4 = {0, 0, 0, 0};
      int32x8 z8 = {0, 0, 0, 0, 0, 0, 0, 0};
      __builtin_amdgcn_tensor_load_to_lds(g0, gd, z4, z4, z8, 0);
    }

    // ---- (2) all threads stage A tile (f32 -> bf16)
    {
      bool fastA = (m0 + 64 <= Mr) && (k0 + 32 <= Kd);
      if (fastA) {
        if (k0 + 32 < Kd)
          __builtin_prefetch(&A[(size_t)(m0 + lrow) * Kd + k0 + 32], 0, 1);
        const float4* src =
            (const float4*)&A[(size_t)(m0 + lrow) * Kd + k0 + lcol];
        float4 v0 = src[0];
        float4 v1 = src[1];
        uint4 pk;
        pk.x = pack2bf(v0.x, v0.y);
        pk.y = pack2bf(v0.z, v0.w);
        pk.z = pack2bf(v1.x, v1.y);
        pk.w = pack2bf(v1.z, v1.w);
        *(uint4*)&As[lrow][lcol] = pk;
      } else {
        int gr = m0 + lrow;
#pragma unroll
        for (int c = 0; c < 8; ++c) {
          int gk = k0 + lcol + c;
          float av = (gr < Mr && gk < Kd) ? A[(size_t)gr * Kd + gk] : 0.0f;
          As[lrow][lcol + c] = f2bf(av);
        }
      }
    }

    // ---- (3) wave0 waits for TDM completion, then publish to the block
    if (wave0) __builtin_amdgcn_s_wait_tensorcnt(0);
    __syncthreads();

    // ---- (4) cooperative pack Ws (f32) -> Bs (bf16)
    {
      const float4* s = (const float4*)&Ws[lrow][lcol];
      float4 v0 = s[0];
      float4 v1 = s[1];
      uint4 pk;
      pk.x = pack2bf(v0.x, v0.y);
      pk.y = pack2bf(v0.z, v0.w);
      pk.z = pack2bf(v1.x, v1.y);
      pk.w = pack2bf(v1.z, v1.w);
      *(uint4*)&Bs[lrow][lcol] = pk;
    }
    __syncthreads();

    // ---- (5) fragments + WMMA
    const int half = lane >> 4;   // which K-half this lane holds
    const int l16  = lane & 15;

    // A fragment, 16x32 bf16 in this code's contiguous k order (not the ISA table order):
    //  lanes 0-15 hold K 0..15, lanes 16-31 hold K 16..31 (contiguous, matches B)
    v8bf alo = *(const v8bf*)&As[wm * 16 + l16][half * 16];
    v8bf ahi = *(const v8bf*)&As[wm * 16 + l16][half * 16 + 8];
    v16bf af;
#pragma unroll
    for (int i = 0; i < 8; ++i) { af[i] = alo[i]; af[8 + i] = ahi[i]; }

    // B fragments: lane n holds W row (n0 + n), contiguous 16 K values per half
    v16bf b0 = *(const v16bf*)&Bs[wn * 32 + l16][half * 16];
    v16bf b1 = *(const v16bf*)&Bs[wn * 32 + 16 + l16][half * 16];

    acc0 = __builtin_amdgcn_wmma_f32_16x16x32_bf16(
        false, af, false, b0, (short)0, acc0, false, false);
    acc1 = __builtin_amdgcn_wmma_f32_16x16x32_bf16(
        false, af, false, b1, (short)0, acc1, false, false);
    __syncthreads();
  }

  // epilogue: C layout = VGPR r -> M = (lane<16 ? r : 8+r), N = lane&15
  const int l16   = lane & 15;
  const int rbase = (lane >> 4) * 8;
  const int col0  = n0 + wn * 32 + l16;
  const int col1  = col0 + 16;
#pragma unroll
  for (int r = 0; r < 8; ++r) {
    int row = m0 + wm * 16 + rbase + r;
    if (row >= Mr) continue;
    if (col0 < Nout) {
      size_t idx = (size_t)row * Nout + col0;
      float v = acc0[r] + bias[col0];
      if (ACCUM) v += C[idx];
      if (RELU) v = fmaxf(v, 0.0f);
      C[idx] = v;
    }
    if (col1 < Nout) {
      size_t idx = (size_t)row * Nout + col1;
      float v = acc1[r] + bias[col1];
      if (ACCUM) v += C[idx];
      if (RELU) v = fmaxf(v, 0.0f);
      C[idx] = v;
    }
  }
}

// ---------------------------------------------------------------------------
// BatchNorm (batch statistics): column sum/sumsq via atomics, then apply.
// ---------------------------------------------------------------------------
__global__ void k_bnstats(const float* __restrict__ X, float* __restrict__ stats,
                          int rows, int cols, int rowsPerChunk) {
  int col = blockIdx.x * blockDim.x + threadIdx.x;
  if (col >= cols) return;
  int r0 = blockIdx.y * rowsPerChunk;
  int r1 = r0 + rowsPerChunk; if (r1 > rows) r1 = rows;
  float s = 0.f, ss = 0.f;
  for (int r = r0; r < r1; ++r) {
    float v = X[(size_t)r * cols + col];
    s += v; ss += v * v;
  }
  atomicAdd(&stats[col], s);
  atomicAdd(&stats[cols + col], ss);
}

__global__ void k_bnapply(const float* __restrict__ X, float* __restrict__ Y,
                          const float* __restrict__ stats,
                          const float* __restrict__ g, const float* __restrict__ b,
                          int rows, int cols, int relu) {
  size_t i = (size_t)blockIdx.x * blockDim.x + threadIdx.x;
  size_t total = (size_t)rows * cols;
  if (i >= total) return;
  int col = (int)(i % cols);
  float inv_n = 1.0f / (float)rows;
  float mu  = stats[col] * inv_n;
  float var = stats[cols + col] * inv_n - mu * mu;
  float y = (X[i] - mu) * rsqrtf(var + 1e-3f) * g[col] + b[col];
  if (relu) y = fmaxf(y, 0.f);
  Y[i] = y;
}

// ---------------------------------------------------------------------------
// Segment softmax over (N,8) with segments `inverse`, then weighted scatter.
// ---------------------------------------------------------------------------
__global__ void k_fill_u32(unsigned* __restrict__ p, unsigned v, size_t n) {
  size_t i = (size_t)blockIdx.x * blockDim.x + threadIdx.x;
  if (i < n) p[i] = v;
}

__global__ void k_relu_ip(float* __restrict__ p, size_t n) {
  size_t i = (size_t)blockIdx.x * blockDim.x + threadIdx.x;
  if (i < n) p[i] = fmaxf(p[i], 0.f);
}

__global__ void k_seg_max(const float* __restrict__ s, const int* __restrict__ inv,
                          unsigned* __restrict__ smax, int N) {
  size_t i = (size_t)blockIdx.x * blockDim.x + threadIdx.x;
  if (i >= (size_t)N * 8) return;
  int n = (int)(i >> 3), j = (int)(i & 7);
  atomicMax(&smax[(size_t)inv[n] * 8 + j], map_ord(s[i]));
}

__global__ void k_seg_expsum(float* __restrict__ s, const unsigned* __restrict__ smax,
                             const int* __restrict__ inv, float* __restrict__ denom,
                             int N) {
  size_t i = (size_t)blockIdx.x * blockDim.x + threadIdx.x;
  if (i >= (size_t)N * 8) return;
  int n = (int)(i >> 3), j = (int)(i & 7);
  float mx = unmap_ord(smax[(size_t)inv[n] * 8 + j]);
  float e = expf(s[i] - mx);
  s[i] = e;  // s becomes e in place
  atomicAdd(&denom[(size_t)inv[n] * 8 + j], e);
}

// dot = (attn ⊗ x) scatter-summed into xseg (M,512); attn = e/denom[inv]
__global__ void k_seg_scatter(const float* __restrict__ e, const float* __restrict__ denom,
                              const float* __restrict__ x, const int* __restrict__ inv,
                              float* __restrict__ xseg, int N) {
  size_t i = (size_t)blockIdx.x * blockDim.x + threadIdx.x;
  if (i >= (size_t)N * 512) return;
  int n = (int)(i >> 9);
  int idx = (int)(i & 511);
  int j = idx >> 6, c = idx & 63;
  int seg = inv[n];
  float a = e[(size_t)n * 8 + j] / denom[(size_t)seg * 8 + j];
  atomicAdd(&xseg[(size_t)seg * 512 + idx], a * x[(size_t)n * 64 + c]);
}

// ---------------------------------------------------------------------------
// Positional embedding (sin/cos, 3 axes x 64 dims -> 192)
// ---------------------------------------------------------------------------
__global__ void k_pe_emb(const float* __restrict__ coords, float* __restrict__ emb,
                         int Mr) {
  int m = blockIdx.x;
  int t = threadIdx.x;        // 96 threads: axis a=t/32, pair p=t%32
  if (m >= Mr || t >= 96) return;
  int a = t / 32, p = t % 32;
  float raw = coords[(size_t)m * 4 + 1 + a];
  float z = raw / (1.0f + 1e-6f) * 6.283185307179586f;
  float dt = powf(10000.0f, 2.0f * (float)p / 64.0f);
  float pos = z / dt;
  emb[(size_t)m * 192 + a * 64 + 2 * p + 0] = sinf(pos);
  emb[(size_t)m * 192 + a * 64 + 2 * p + 1] = cosf(pos);
}

// h = relu(score[:, :, None] * x[:, None, :])  ->  (rows*8, d)
__global__ void k_outer_h(const float* __restrict__ sc, const float* __restrict__ x,
                          float* __restrict__ h, int rows, int d) {
  size_t i = (size_t)blockIdx.x * blockDim.x + threadIdx.x;
  size_t total = (size_t)rows * 8 * d;
  if (i >= total) return;
  int c = (int)(i % d);
  size_t mj = i / d;
  int j = (int)(mj & 7);
  int m = (int)(mj >> 3);
  float v = sc[(size_t)m * 8 + j] * x[(size_t)m * d + c];
  h[i] = fmaxf(v, 0.f);
}

// ---------------------------------------------------------------------------
// Tiny per-row attention: 1 query, 8 keys of dim d. One block (64 thr) per row.
// ---------------------------------------------------------------------------
__global__ void k_attn8(const float* __restrict__ q, const float* __restrict__ km,
                        const float* __restrict__ vm, float* __restrict__ out,
                        int rows, int d) {
  int row = blockIdx.x;
  if (row >= rows) return;
  int tid = threadIdx.x;
  __shared__ float sw[8];
  if (tid < 8) {
    const float* qp = q + (size_t)row * d;
    const float* kp = km + ((size_t)row * 8 + tid) * d;
    float acc = 0.f;
    for (int c = 0; c < d; ++c) acc += qp[c] * kp[c];
    sw[tid] = acc * rsqrtf((float)d);
  }
  __syncthreads();
  if (tid == 0) {
    float mx = -1e30f;
    for (int j = 0; j < 8; ++j) mx = fmaxf(mx, sw[j]);
    float s = 0.f, w[8];
    for (int j = 0; j < 8; ++j) { w[j] = expf(sw[j] - mx); s += w[j]; }
    float inv = 1.0f / s;
    for (int j = 0; j < 8; ++j) sw[j] = w[j] * inv;
  }
  __syncthreads();
  for (int c = tid; c < d; c += 64) {
    float acc = 0.f;
#pragma unroll
    for (int j = 0; j < 8; ++j)
      acc += sw[j] * vm[((size_t)row * 8 + j) * d + c];
    out[(size_t)row * d + c] = acc;
  }
}

__global__ void k_gather(const float* __restrict__ src, const int* __restrict__ inv,
                         float* __restrict__ dst, int N, int D) {
  size_t i = (size_t)blockIdx.x * blockDim.x + threadIdx.x;
  if (i >= (size_t)N * D) return;
  int n = (int)(i / D), c = (int)(i % D);
  dst[i] = src[(size_t)inv[n] * D + c];
}

__global__ void k_concat2(const float* __restrict__ a, const float* __restrict__ b,
                          float* __restrict__ dst, int rows, int da, int db) {
  size_t i = (size_t)blockIdx.x * blockDim.x + threadIdx.x;
  int w = da + db;
  if (i >= (size_t)rows * w) return;
  int r = (int)(i / w), c = (int)(i % w);
  dst[i] = (c < da) ? a[(size_t)r * da + c] : b[(size_t)r * db + (c - da)];
}

// ---------------------------------------------------------------------------
// Host orchestration
// ---------------------------------------------------------------------------
struct LinP { const float *W, *b; };
struct BnP  { const float *g, *b; };
struct VsaP {
  LinP l1; BnP bn1; LinP l2; BnP bn2; LinP l3; BnP bn3;
  LinP score; BnP norm; LinP q, k, v, o;
};

static inline unsigned g1(size_t total, int bs) {
  return (unsigned)((total + (size_t)bs - 1) / (size_t)bs);
}

static void gemm(hipStream_t st, const float* A, LinP w, float* C,
                 int Mr, int Nout, int Kd, bool relu, bool accum) {
  dim3 grid((Nout + 63) / 64, (Mr + 63) / 64), blk(256);
  if (!relu && !accum)      k_gemm<0, 0><<<grid, blk, 0, st>>>(A, w.W, w.b, C, Mr, Nout, Kd);
  else if (relu && !accum)  k_gemm<1, 0><<<grid, blk, 0, st>>>(A, w.W, w.b, C, Mr, Nout, Kd);
  else if (!relu && accum)  k_gemm<0, 1><<<grid, blk, 0, st>>>(A, w.W, w.b, C, Mr, Nout, Kd);
  else                      k_gemm<1, 1><<<grid, blk, 0, st>>>(A, w.W, w.b, C, Mr, Nout, Kd);
}

static void bnorm(hipStream_t st, const float* X, float* Y, BnP p,
                  size_t rows, int cols, bool relu, float* stats) {
  (void)hipMemsetAsync(stats, 0, (size_t)2 * cols * sizeof(float), st);
  int rpc = (int)((rows + 63) / 64);
  dim3 gs((cols + 255) / 256, 64);
  k_bnstats<<<gs, 256, 0, st>>>(X, stats, (int)rows, cols, rpc);
  size_t total = rows * (size_t)cols;
  k_bnapply<<<dim3(g1(total, 256)), 256, 0, st>>>(X, Y, stats, p.g, p.b,
                                                  (int)rows, cols, relu ? 1 : 0);
}

extern "C" void kernel_launch(void* const* d_in, const int* in_sizes, int n_in,
                              void* d_out, int out_size, void* d_ws, size_t ws_size,
                              hipStream_t stream) {
  (void)in_sizes; (void)out_size; (void)ws_size;
  constexpr int N = 150000, M = 15000;
  if (n_in < 148) return;

  const float* inp    = (const float*)d_in[0];
  const int*   inv    = (const int*)d_in[1];
  const float* coords = (const float*)d_in[2];
  // d_in[3] = bev_shape scalar (unused)

  // --- params walker (setup_inputs insertion order) ---
  int pi = 4;
  auto nxt = [&]() { return (const float*)d_in[pi++]; };
  auto L = [&]() { LinP l; l.W = nxt(); l.b = nxt(); return l; };
  auto B = [&]() { BnP x; x.g = nxt(); x.b = nxt(); return x; };
  auto V = [&]() {
    VsaP v;
    v.l1 = L(); v.bn1 = B(); v.l2 = L(); v.bn2 = B(); v.l3 = L(); v.bn3 = B();
    v.score = L(); v.norm = B();
    v.q = L(); v.k = L(); v.v = L(); v.o = L();
    return v;
  };
  VsaP outer = V();
  LinP embed1 = L(), embed2 = L();
  LinP pe[4]; for (int i = 0; i < 4; ++i) pe[i] = L();
  VsaP lay[4]; for (int i = 0; i < 4; ++i) lay[i] = V();
  LinP post1 = L(); BnP postbn1 = B();
  LinP post2 = L(); BnP postbn2 = B();
  LinP post3 = L(); BnP postbn3 = B();

  // --- workspace layout ---
  char* base = (char*)d_ws;
  size_t off = 0;
  auto alloc = [&](size_t elems) {
    float* p = (float*)(base + off);
    off += ((elems * 4 + 255) & ~(size_t)255);
    return p;
  };
  float* xa    = alloc((size_t)N * 64);
  float* xb    = alloc((size_t)N * 64);
  float* sc    = alloc((size_t)N * 8);
  float* smax  = alloc((size_t)M * 8);
  float* denom = alloc((size_t)M * 8);
  float* xseg  = alloc((size_t)M * 512);
  float* stats = alloc(2048);
  float* emb   = alloc((size_t)M * 192);
  float* srcA  = alloc((size_t)M * 1024);
  float* srcB  = alloc((size_t)M * 1024);
  float* t0    = alloc((size_t)M * 512);
  float* t1    = alloc((size_t)M * 512);
  float* scL   = alloc((size_t)M * 8);
  float* hb    = alloc((size_t)M * 4096);
  float* qb    = alloc((size_t)M * 512);
  float* kb    = alloc((size_t)M * 4096);
  float* vb    = alloc((size_t)M * 4096);
  float* ao    = alloc((size_t)M * 512);
  float* ob    = alloc((size_t)M * 512);
  float* hg    = alloc((size_t)N * 512);
  float* kN    = alloc((size_t)N * 512);
  float* vN    = alloc((size_t)N * 512);
  float* qN    = alloc((size_t)N * 64);
  float* aoN   = alloc((size_t)N * 64);
  float* oN    = alloc((size_t)N * 64);

  // ---- outer pre_mlp on N points: x = bn3(l3(relu(bn2(l2(relu(bn1(l1)))))))
  gemm(stream, inp, outer.l1, xa, N, 64, 64, false, false);
  bnorm(stream, xa, xb, outer.bn1, N, 64, true, stats);
  gemm(stream, xb, outer.l2, xa, N, 64, 64, false, false);
  bnorm(stream, xa, xb, outer.bn2, N, 64, true, stats);
  gemm(stream, xb, outer.l3, xa, N, 64, 64, false, false);
  bnorm(stream, xa, xb, outer.bn3, N, 64, false, stats);   // x = xb
  gemm(stream, xb, outer.score, sc, N, 8, 64, false, false);

  // ---- segment softmax + weighted scatter into voxel features
  k_fill_u32<<<dim3(g1((size_t)M * 8, 256)), 256, 0, stream>>>(
      (unsigned*)smax, 0x007FFFFFu /* map_ord(-inf) */, (size_t)M * 8);
  (void)hipMemsetAsync(denom, 0, (size_t)M * 8 * sizeof(float), stream);
  (void)hipMemsetAsync(xseg, 0, (size_t)M * 512 * sizeof(float), stream);
  k_seg_max<<<dim3(g1((size_t)N * 8, 256)), 256, 0, stream>>>(sc, inv, (unsigned*)smax, N);
  k_seg_expsum<<<dim3(g1((size_t)N * 8, 256)), 256, 0, stream>>>(sc, (const unsigned*)smax, inv, denom, N);
  k_seg_scatter<<<dim3(g1((size_t)N * 512, 256)), 256, 0, stream>>>(sc, denom, xb, inv, xseg, N);
  k_relu_ip<<<dim3(g1((size_t)M * 512, 256)), 256, 0, stream>>>(xseg, (size_t)M * 512);

  // ---- tsfm_ffn: embed 512 -> 16 (relu) -> 64
  gemm(stream, xseg, embed1, t0, M, 16, 512, true, false);
  gemm(stream, t0, embed2, srcA, M, 64, 16, false, false);
  k_pe_emb<<<dim3(M), dim3(96), 0, stream>>>(coords, emb, M);

  float* src = srcA;
  float* dst = srcB;
  for (int i = 0; i < 4; ++i) {
    int d = 64 << i;
    VsaP& P = lay[i];
    // src += pef (GEMM with accumulate epilogue)
    gemm(stream, emb, pe[i], src, M, d, 192, false, true);
    // pre_mlp
    gemm(stream, src, P.l1, t0, M, d, d, false, false);
    bnorm(stream, t0, t1, P.bn1, M, d, true, stats);
    gemm(stream, t1, P.l2, t0, M, d, d, false, false);
    bnorm(stream, t0, t1, P.bn2, M, d, true, stats);
    gemm(stream, t1, P.l3, t0, M, d, d, false, false);
    bnorm(stream, t0, t1, P.bn3, M, d, false, stats);      // x = t1
    // score + h + norm
    gemm(stream, t1, P.score, scL, M, 8, d, false, false);
    k_outer_h<<<dim3(g1((size_t)M * 8 * d, 256)), 256, 0, stream>>>(scL, t1, hb, M, d);
    bnorm(stream, hb, hb, P.norm, (size_t)M * 8, d, false, stats);
    // mha (q from x; k,v from hs on M*8 rows)
    gemm(stream, t1, P.q, qb, M, d, d, false, false);
    gemm(stream, hb, P.k, kb, M * 8, d, d, false, false);
    gemm(stream, hb, P.v, vb, M * 8, d, d, false, false);
    k_attn8<<<dim3(M), dim3(64), 0, stream>>>(qb, kb, vb, ao, M, d);
    gemm(stream, ao, P.o, ob, M, d, d, false, false);
    // concat [src, o] -> width 2d
    k_concat2<<<dim3(g1((size_t)M * 2 * d, 256)), 256, 0, stream>>>(src, ob, dst, M, d, d);
    float* tmp = src; src = dst; dst = tmp;
  }

  // ---- post MLP: 1024 -> 512 (bn relu) -> 512 (bn relu) -> 512 (bn)
  gemm(stream, src, post1, t0, M, 512, 1024, false, false);
  bnorm(stream, t0, t1, postbn1, M, 512, true, stats);
  gemm(stream, t1, post2, t0, M, 512, 512, false, false);
  bnorm(stream, t0, t1, postbn2, M, 512, true, stats);
  gemm(stream, t1, post3, t0, M, 512, 512, false, false);
  bnorm(stream, t0, t1, postbn3, M, 512, false, stats);    // h = t1 (M,512)

  // ---- outer attention: gather h[inverse], bn(norm) over (N*8,64), mha
  k_gather<<<dim3(g1((size_t)N * 512, 256)), 256, 0, stream>>>(t1, inv, hg, N, 512);
  bnorm(stream, hg, hg, outer.norm, (size_t)N * 8, 64, false, stats);
  gemm(stream, xb, outer.q, qN, N, 64, 64, false, false);
  gemm(stream, hg, outer.k, kN, N * 8, 64, 64, false, false);
  gemm(stream, hg, outer.v, vN, N * 8, 64, 64, false, false);
  k_attn8<<<dim3(N), dim3(64), 0, stream>>>(qN, kN, vN, aoN, N, 64);
  gemm(stream, aoN, outer.o, oN, N, 64, 64, false, false);

  // ---- out = concat([inp, o], -1)  (N x 128, f32)
  k_concat2<<<dim3(g1((size_t)N * 128, 256)), 256, 0, stream>>>(inp, oN, (float*)d_out, N, 64, 64);
}
